// ModConv2D_17334488006733
// MI455X (gfx1250) — compile-verified, matrix operand order checked by reading, not run on hardware
//
#include <hip/hip_runtime.h>
#include <hip/hip_bf16.h>
#include <cstdint>
#include <cstddef>

// ---------------------------------------------------------------------------
// ModulatedConv2D (StyleGAN2) for MI455X / gfx1250.
//   x:      [8, 512, 64, 64] f32
//   style:  [8, 512]         f32
//   weight: [512, 512, 3, 3] f32
//   bias:   [512]            f32
//   out:    [8, 512, 64, 64] f32
// Strategy: implicit GEMM per sample with v_wmma_f32_16x16x32_bf16.
//   M = OUT(512), N = H*W(4096), K = IN*9(4608)
// ---------------------------------------------------------------------------

#define BATCH   8
#define IN_DIM  512
#define OUT_DIM 512
#define HW      64
#define PAD_HW  66
#define KTAPS   9

#define EQ_C   0.014731391f          // 1/sqrt(512*9)
#define EQ_C2  (1.0f/4608.0f)        // EQ_C^2
#define EPS    1e-8f

typedef __attribute__((ext_vector_type(16))) __bf16 v16bf;
typedef __attribute__((ext_vector_type(8)))  float  v8f;

__device__ __forceinline__ unsigned short f32_to_bf16_rte(float f) {
    unsigned int u = __float_as_uint(f);
    u += 0x7FFFu + ((u >> 16) & 1u);     // round-to-nearest-even
    return (unsigned short)(u >> 16);
}

// ---------------------------------------------------------------------------
// Kernel 1: sigma_inv[b][o] = rsqrt(EQ_C^2 * sum_i (s_bi+1)^2 * sum_kk W^2 + eps)
// One block per (o, b), 256 threads, LDS tree reduction.
// ---------------------------------------------------------------------------
__global__ __launch_bounds__(256)
void sigma_kernel(const float* __restrict__ weight,
                  const float* __restrict__ style,
                  float* __restrict__ sigma_inv) {
    __shared__ float red[256];
    const int o = blockIdx.x;
    const int b = blockIdx.y;
    const int t = threadIdx.x;

    float sum = 0.0f;
    for (int i = t; i < IN_DIM; i += 256) {
        const float* wp = weight + ((size_t)o * IN_DIM + i) * KTAPS;
        float wsq = 0.0f;
#pragma unroll
        for (int k = 0; k < KTAPS; ++k) { float v = wp[k]; wsq += v * v; }
        float s = style[b * IN_DIM + i] + 1.0f;
        sum += wsq * s * s;
    }
    red[t] = sum;
    __syncthreads();
#pragma unroll
    for (int off = 128; off > 0; off >>= 1) {
        if (t < off) red[t] += red[t + off];
        __syncthreads();
    }
    if (t == 0)
        sigma_inv[b * OUT_DIM + o] = rsqrtf(red[0] * EQ_C2 + EPS);
}

// ---------------------------------------------------------------------------
// Kernel 2: modulated bf16 weights, layout wmod[b][tap][o][i] (i contiguous)
// ---------------------------------------------------------------------------
__global__ __launch_bounds__(256)
void wmod_kernel(const float* __restrict__ weight,
                 const float* __restrict__ style,
                 const float* __restrict__ sigma_inv,
                 unsigned short* __restrict__ wmod) {
    const int e = blockIdx.x * 256 + threadIdx.x;   // e = ((b*9+t)*512+o)*512+i
    const int total = BATCH * KTAPS * OUT_DIM * IN_DIM;
    if (e >= total) return;
    const int i  = e & 511;
    const int o  = (e >> 9) & 511;
    const int bt = e >> 18;          // b*9 + t
    const int t  = bt % 9;
    const int b  = bt / 9;
    float v = weight[((size_t)o * IN_DIM + i) * KTAPS + t] * EQ_C
            * (style[b * IN_DIM + i] + 1.0f)
            * sigma_inv[b * OUT_DIM + o];
    wmod[e] = f32_to_bf16_rte(v);
}

// ---------------------------------------------------------------------------
// Kernel 3: x NCHW f32 -> padded NHWC bf16 (halo pre-zeroed by memset).
// LDS-tiled 64ch x 64w transpose; one block per (b, h-row, c-tile of 64).
// ---------------------------------------------------------------------------
__global__ __launch_bounds__(256)
void xprep_kernel(const float* __restrict__ x, unsigned short* __restrict__ xT) {
    __shared__ unsigned short lds[64 * 65];   // [c][w], padded stride
    const int id = blockIdx.x;
    const int ct = id & 7;           // channel tile
    const int hh = (id >> 3) & 63;   // image row
    const int b  = id >> 9;
    const int t  = threadIdx.x;
    const int c0 = ct * 64;

    {   // read coalesced along w
        const int w  = t & 63;
        const int cl = t >> 6;       // 0..3
#pragma unroll
        for (int r = 0; r < 16; ++r) {
            const int c = cl + r * 4;
            float v = x[(((size_t)b * IN_DIM + c0 + c) * HW + hh) * HW + w];
            lds[c * 65 + w] = f32_to_bf16_rte(v);
        }
    }
    __syncthreads();
    {   // write coalesced along c
        const int c  = t & 63;
        const int wl = t >> 6;
#pragma unroll
        for (int r = 0; r < 16; ++r) {
            const int w = wl + r * 4;
            xT[(((size_t)b * PAD_HW + hh + 1) * PAD_HW + (w + 1)) * IN_DIM + c0 + c]
                = lds[c * 65 + w];
        }
    }
}

// ---------------------------------------------------------------------------
// Kernel 4: implicit-GEMM conv with WMMA bf16.
// Block = 256 threads = 8 wave32s. Each wave: 64 outs x 32 pixels
//   -> 4 M-tiles x 2 N-tiles = 8 f32 accumulators (v8f each).
// grid.x = 128 pixel tiles per sample (h in 0..63, w0 in {0,32}); grid.y = b.
// K loop: 16 ci-steps of 32 channels x 9 taps -> 144 x 8 WMMAs per wave.
// ---------------------------------------------------------------------------
__global__ __launch_bounds__(256)
void conv_kernel(const unsigned short* __restrict__ xTu,
                 const unsigned short* __restrict__ wmodu,
                 const float* __restrict__ bias,
                 float* __restrict__ out) {
    const __bf16* __restrict__ xT   = (const __bf16*)xTu;
    const __bf16* __restrict__ wmod = (const __bf16*)wmodu;

    const int b    = blockIdx.y;
    const int tile = blockIdx.x;
    const int h    = tile >> 1;
    const int w0   = (tile & 1) << 5;        // 0 or 32
    const int wave = threadIdx.x >> 5;
    const int lane = threadIdx.x & 31;
    const int col   = lane & 15;             // A-row-in-lane / B-pixel-in-lane
    const int khalf = lane >> 4;             // K half: lanes 0-15 -> K 0..15
    const int obase = wave << 6;             // 64 output channels per wave

    const __bf16* __restrict__ xb = xT + (size_t)b * (PAD_HW * PAD_HW * IN_DIM);

    v8f acc[8] = {{}, {}, {}, {}, {}, {}, {}, {}};

    for (int ci = 0; ci < IN_DIM; ci += 32) {
        // prefetch next ci block of weights (global_prefetch_b8)
        __builtin_prefetch(wmod + ((size_t)(b * KTAPS * OUT_DIM + obase + col)) * IN_DIM
                                + ci + 32, 0, 1);
#pragma unroll
        for (int t = 0; t < KTAPS; ++t) {
            const int dh = t / 3 - 1;
            const int dw = t % 3 - 1;
            // B fragments: 32 in-channels x 16 pixels, padded NHWC -> contiguous
            const __bf16* bp = xb + (size_t)((h + 1 + dh) * PAD_HW + (w0 + 1 + dw + col)) * IN_DIM
                                  + ci + (khalf << 4);
            v16bf bf0 = *(const v16bf*)bp;
            v16bf bf1 = *(const v16bf*)(bp + 16 * IN_DIM);   // pixels +16

            const __bf16* wp = wmod + ((size_t)((b * KTAPS + t) * OUT_DIM + obase + col)) * IN_DIM
                                    + ci + (khalf << 4);
#pragma unroll
            for (int m = 0; m < 4; ++m) {
                v16bf a = *(const v16bf*)(wp + (size_t)m * 16 * IN_DIM);
                acc[2 * m + 0] = __builtin_amdgcn_wmma_f32_16x16x32_bf16(
                    false, a, false, bf0, (short)0, acc[2 * m + 0], false, false);
                acc[2 * m + 1] = __builtin_amdgcn_wmma_f32_16x16x32_bf16(
                    false, a, false, bf1, (short)0, acc[2 * m + 1], false, false);
            }
        }
    }

    // C/D layout (ISA 7.12.2): VGPR j, lanes 0-15 -> M=j, lanes 16-31 -> M=j+8
#pragma unroll
    for (int m = 0; m < 4; ++m) {
#pragma unroll
        for (int n = 0; n < 2; ++n) {
            v8f v = acc[2 * m + n];
            const int w = w0 + n * 16 + col;
#pragma unroll
            for (int j = 0; j < 8; ++j) {
                const int o = obase + m * 16 + khalf * 8 + j;
                out[(((size_t)b * OUT_DIM + o) * HW + h) * HW + w] = v[j] + bias[o];
            }
        }
    }
}

// ---------------------------------------------------------------------------
// Launch
// ---------------------------------------------------------------------------
extern "C" void kernel_launch(void* const* d_in, const int* in_sizes, int n_in,
                              void* d_out, int out_size, void* d_ws, size_t ws_size,
                              hipStream_t stream) {
    const float* x      = (const float*)d_in[0];
    const float* style  = (const float*)d_in[1];
    const float* weight = (const float*)d_in[2];
    const float* bias   = (const float*)d_in[3];
    float* out = (float*)d_out;

    // Workspace layout (all 32B-aligned):
    //   sigma_inv : 8*512 f32                      =     16,384 B
    //   xT        : 8*66*66*512 bf16 (padded NHWC) = 35,684,352 B
    //   wmod      : 8*9*512*512 bf16               = 37,748,736 B
    char* ws = (char*)d_ws;
    float* sigma_inv = (float*)ws;
    const size_t xT_off   = 16384;
    const size_t xT_bytes = (size_t)BATCH * PAD_HW * PAD_HW * IN_DIM * sizeof(unsigned short);
    unsigned short* xT   = (unsigned short*)(ws + xT_off);
    unsigned short* wmod = (unsigned short*)(ws + xT_off + xT_bytes);

    // zero padded x buffer (halo stays 0 -> implicit conv padding)
    hipMemsetAsync(xT, 0, xT_bytes, stream);

    sigma_kernel<<<dim3(OUT_DIM, BATCH), 256, 0, stream>>>(weight, style, sigma_inv);

    const int wmod_elems = BATCH * KTAPS * OUT_DIM * IN_DIM;   // 18,874,368
    wmod_kernel<<<(wmod_elems + 255) / 256, 256, 0, stream>>>(weight, style, sigma_inv, wmod);

    xprep_kernel<<<BATCH * HW * (IN_DIM / 64), 256, 0, stream>>>(x, xT);

    conv_kernel<<<dim3(128, BATCH), 256, 0, stream>>>(xT, wmod, bias, out);
}
